// fwd_mpgnn_40458591929248
// MI455X (gfx1250) — matrix operand order repaired and checked by reading, not compiled, not run on hardware
//
#include <hip/hip_runtime.h>
#include <math.h>
#include <stdint.h>

#define PP    32768
#define HH    256
#define LL    8
#define FEATN 32
#define HALFP (PP / 2)

// macro-tile: 128 (M) x 64 (N) per 256-thread block; K staged 64-wide.
#define TM 128
#define TN 64
#define KS 64

typedef _Float16 v16h __attribute__((ext_vector_type(16)));
typedef _Float16 v8h  __attribute__((ext_vector_type(8)));
typedef float    v8f  __attribute__((ext_vector_type(8)));

// ---------------------------------------------------------------------------
// CDNA5 async copy: global -> LDS, 16B per lane, tracked by ASYNCcnt.
// VDST VGPR holds the LDS byte address (low 32 bits of the generic pointer:
// generic shared addr = {shared_aperture, lds_offset} per ISA 10.2).
// ---------------------------------------------------------------------------
__device__ __forceinline__ void async_b128(unsigned lds_off, const _Float16* g) {
  asm volatile("global_load_async_to_lds_b128 %0, %1, off"
               :: "v"(lds_off), "v"((unsigned long long)(uintptr_t)g)
               : "memory");
}
__device__ __forceinline__ void wait_async_le6() {
  asm volatile("s_wait_asynccnt 0x6" ::: "memory");
}
__device__ __forceinline__ void wait_async_0() {
  asm volatile("s_wait_asynccnt 0x0" ::: "memory");
}

// CDNA5 hardware tanh (TRANS unit). Trailing v_nop per the ISA trans-op
// hazard rule (one independent op / V_NOP before the result may be consumed),
// since the compiler cannot schedule around inline asm.
__device__ __forceinline__ float hw_tanh(float x) {
  float y;
  asm volatile("v_tanh_f32 %0, %1\n\tv_nop" : "=v"(y) : "v"(x));
  return y;
}

// ---------------------------------------------------------------------------
// Stage copies. LDS slabs have a fixed row stride of KS=64 halves (128 B).
// A slab: TM x kw halves; B slab: TN x kw halves (W stored (N,K) row-major,
// so B column n == W row n: contiguous).  kw in {32, 64}.
// Per-wave async issues for kw=64: A=4, B=2 (total 6); for kw=32: 2+1.
// ---------------------------------------------------------------------------
__device__ __forceinline__ void copy_stage(const _Float16* __restrict__ A, int lda, int rB,
                                           const _Float16* __restrict__ W, int K, int cB,
                                           int k0, int kw,
                                           unsigned aBase, unsigned bBase, int t) {
  int psh   = (kw == 64) ? 3 : 2;       // parts-per-row shift (kw/8 b128 segs)
  int pmask = (1 << psh) - 1;
  int aIter = (TM << psh) >> 8;         // (TM*parts)/256 : 4 or 2
  int bIter = (TN << psh) >> 8;         // 2 or 1
  for (int i = 0; i < aIter; ++i) {
    int seg  = t + (i << 8);
    int row  = seg >> psh, part = seg & pmask;
    async_b128(aBase + (unsigned)(row * (KS * 2) + part * 16),
               A + (size_t)(rB + row) * lda + k0 + part * 8);
  }
  for (int i = 0; i < bIter; ++i) {
    int seg  = t + (i << 8);
    int row  = seg >> psh, part = seg & pmask;
    async_b128(bBase + (unsigned)(row * (KS * 2) + part * 16),
               W + (size_t)(cB + row) * K + k0 + part * 8);
  }
}

// ---------------------------------------------------------------------------
// WMMA fragment loads (CDNA5 16-bit layouts, ISA 7.12.2).
// A 16x32: lanes 0-15 row=lane, K={0..7,16..23}; lanes 16-31 K={8..15,24..31}.
// B 32x16: lanes 0-15 col=lane, K={0..7,16..23}; lanes 16-31 K={8..15,24..31}.
// ---------------------------------------------------------------------------
__device__ __forceinline__ v16h lds_a_frag(const _Float16* base, int r0l, int ks, int lane) {
  int m  = lane & 15;
  int ko = (lane >> 4) << 3;
  const _Float16* p = base + (r0l + m) * KS + ks + ko;
  v8h lo = *(const v8h*)p;
  v8h hi = *(const v8h*)(p + 16);
  v16h r;
#pragma unroll
  for (int i = 0; i < 8; ++i) { r[i] = lo[i]; r[i + 8] = hi[i]; }
  return r;
}
__device__ __forceinline__ v16h lds_b_frag(const _Float16* base, int c0l, int ks, int lane) {
  int n  = lane & 15;
  int ko = (lane >> 4) << 3;
  const _Float16* p = base + (c0l + n) * KS + ks + ko;
  v8h lo = *(const v8h*)p;
  v8h hi = *(const v8h*)(p + 16);
  v16h r;
#pragma unroll
  for (int i = 0; i < 8; ++i) { r[i] = lo[i]; r[i + 8] = hi[i]; }
  return r;
}
__device__ __forceinline__ v16h glb_a_frag(const _Float16* __restrict__ A,
                                           int lda, int r0, int k0, int lane) {
  int m  = lane & 15;
  int ko = (lane >> 4) << 3;
  const _Float16* p = A + (size_t)(r0 + m) * lda + k0 + ko;
  v8h lo = *(const v8h*)p;
  v8h hi = *(const v8h*)(p + 16);
  v16h r;
#pragma unroll
  for (int i = 0; i < 8; ++i) { r[i] = lo[i]; r[i + 8] = hi[i]; }
  return r;
}

// C/D: lanes 0-15: VGPR v -> (M=r0+v, N=c0+lane); lanes 16-31: M=r0+8+v.
__device__ __forceinline__ void epilogue(v8f acc, int r0, int c0, int lane,
                                         const float* __restrict__ bias,
                                         _Float16* __restrict__ o16, int ldo,
                                         float* __restrict__ o32, int ldo32,
                                         int doTanh) {
  int n  = c0 + (lane & 15);
  int rb = r0 + ((lane >> 4) << 3);
  float b = bias[n];
#pragma unroll
  for (int v = 0; v < 8; ++v) {
    float x = acc[v] + b;
    if (doTanh) x = hw_tanh(x);
    int r = rb + v;
    o16[(size_t)r * ldo + n] = (_Float16)x;
    if (o32) o32[(size_t)r * ldo32 + n] = x;
  }
}

// ---------------------------------------------------------------------------
// Fused GEMM: out = act( (A [+ A2]) @ W^T + bias ), W stored (N,K) row-major.
// 8 waves / 128x64 macro-tile, LDS double-buffered via async copies.
// ---------------------------------------------------------------------------
template <bool RES>
__global__ __launch_bounds__(256) void wmma_gemm_lds_kernel(
    const _Float16* __restrict__ A,  int lda,
    const _Float16* __restrict__ A2, int lda2,
    const _Float16* __restrict__ W,
    const float*    __restrict__ bias,
    _Float16* __restrict__ o16, int ldo,
    float*    __restrict__ o32, int ldo32,
    int M, int N, int K, int doTanh)
{
  __shared__ __align__(64) _Float16 As[2][TM * KS];   // 2 x 16 KB
  __shared__ __align__(64) _Float16 Bs[2][TN * KS];   // 2 x  8 KB

  int t    = threadIdx.x;
  int lane = t & 31;
  int wid  = t >> 5;
  int tilesN = N / TN;
  int rB = (blockIdx.x / tilesN) * TM;
  int cB = (blockIdx.x % tilesN) * TN;
  int r0 = (wid & 3) << 5;   // wave sub-tile (local): 4 x 2 waves of 32x32
  int c0 = (wid >> 2) << 5;

  unsigned aB0 = (unsigned)(uintptr_t)&As[0][0];
  unsigned aB1 = (unsigned)(uintptr_t)&As[1][0];
  unsigned bB0 = (unsigned)(uintptr_t)&Bs[0][0];
  unsigned bB1 = (unsigned)(uintptr_t)&Bs[1][0];

  v8f c00 = {}, c01 = {}, c10 = {}, c11 = {};

  int nst = (K + KS - 1) / KS;   // K in {32,256,512}: tail only when K==32
  copy_stage(A, lda, rB, W, K, cB, 0, (K < KS) ? 32 : KS, aB0, bB0, t);

  for (int s = 0; s < nst; ++s) {
    int cur = s & 1;
    int k0  = s * KS;
    if (s + 1 < nst) {
      copy_stage(A, lda, rB, W, K, cB, k0 + KS, KS,
                 cur ? aB0 : aB1, cur ? bB0 : bB1, t);
      wait_async_le6();          // current stage landed; next stage in flight
    } else {
      wait_async_0();
    }
    __syncthreads();

    const _Float16* Ac = As[cur];
    const _Float16* Bc = Bs[cur];
#pragma unroll
    for (int ks = 0; ks < KS; ks += 32) {
      if (ks && K - k0 <= 32) break;   // K==32 tail stage: single sub-step
      v16h a0 = lds_a_frag(Ac, r0,      ks, lane);
      v16h a1 = lds_a_frag(Ac, r0 + 16, ks, lane);
      if constexpr (RES) {             // resnet residual: (h2 + x) before lin3
        a0 += glb_a_frag(A2, lda2, rB + r0,      k0 + ks, lane);
        a1 += glb_a_frag(A2, lda2, rB + r0 + 16, k0 + ks, lane);
      }
      v16h b0 = lds_b_frag(Bc, c0,      ks, lane);
      v16h b1 = lds_b_frag(Bc, c0 + 16, ks, lane);
      c00 = __builtin_amdgcn_wmma_f32_16x16x32_f16(false, a0, false, b0, (short)0, c00, false, false);
      c01 = __builtin_amdgcn_wmma_f32_16x16x32_f16(false, a0, false, b1, (short)0, c01, false, false);
      c10 = __builtin_amdgcn_wmma_f32_16x16x32_f16(false, a1, false, b0, (short)0, c10, false, false);
      c11 = __builtin_amdgcn_wmma_f32_16x16x32_f16(false, a1, false, b1, (short)0, c11, false, false);
    }
    __syncthreads();                   // all reads done before buffer reuse
  }

  epilogue(c00, rB + r0,      cB + c0,      lane, bias, o16, ldo, o32, ldo32, doTanh);
  epilogue(c01, rB + r0,      cB + c0 + 16, lane, bias, o16, ldo, o32, ldo32, doTanh);
  epilogue(c10, rB + r0 + 16, cB + c0,      lane, bias, o16, ldo, o32, ldo32, doTanh);
  epilogue(c11, rB + r0 + 16, cB + c0 + 16, lane, bias, o16, ldo, o32, ldo32, doTanh);
}

// ---------------------------------------------------------------------------
// Elementwise helpers
// ---------------------------------------------------------------------------
__global__ void cvt_kernel(const float* __restrict__ s, _Float16* __restrict__ d, int n) {
  int t = blockIdx.x * blockDim.x + threadIdx.x;
  if (t < n) d[t] = (_Float16)s[t];
}

__global__ void gather_u_kernel(const _Float16* __restrict__ prev,
                                const int* __restrict__ pb,
                                _Float16* __restrict__ dst) {
  int t = blockIdx.x * blockDim.x + threadIdx.x;
  if (t >= HALFP * HH) return;
  int i = t >> 8, c = t & 255;
  int p0 = pb[(size_t)i * 2 + 0];
  dst[(size_t)i * HH + c] = prev[(size_t)p0 * HH + c];
}

__global__ void gather_b_kernel(const _Float16* __restrict__ prev,
                                const int* __restrict__ pb,
                                _Float16* __restrict__ dst) {
  int t = blockIdx.x * blockDim.x + threadIdx.x;
  if (t >= HALFP * HH) return;
  int i = t >> 8, c = t & 255;
  int p0 = pb[(size_t)(HALFP + i) * 2 + 0];
  int p1 = pb[(size_t)(HALFP + i) * 2 + 1];
  dst[(size_t)i * 512 + c]       = prev[(size_t)p0 * HH + c];
  dst[(size_t)i * 512 + 256 + c] = prev[(size_t)p1 * HH + c];
}

__global__ void concat_hi_kernel(const _Float16* __restrict__ src, _Float16* __restrict__ dst) {
  int t = blockIdx.x * blockDim.x + threadIdx.x;
  if (t >= PP * HH) return;
  int i = t >> 8, c = t & 255;
  dst[(size_t)i * 512 + 256 + c] = src[t];
}

__global__ void zero_hi_kernel(_Float16* __restrict__ dst) {
  int t = blockIdx.x * blockDim.x + threadIdx.x;
  if (t >= PP * HH) return;
  int i = t >> 8, c = t & 255;
  dst[(size_t)i * 512 + 256 + c] = (_Float16)0.0f;
}

// ---------------------------------------------------------------------------
// Host orchestration
// ---------------------------------------------------------------------------
struct LinW { const _Float16* w; const float* b; };
struct ResW { LinW l1, l2, l3; int din, dh, dout; };

static inline void launch_gemm(hipStream_t s,
                               const _Float16* A, int lda,
                               const _Float16* A2, int lda2,
                               const _Float16* W, const float* bias,
                               _Float16* o16, int ldo,
                               float* o32, int ldo32,
                               int M, int N, int K, int act) {
  int blocks = (M / TM) * (N / TN);
  if (A2)
    wmma_gemm_lds_kernel<true><<<blocks, 256, 0, s>>>(A, lda, A2, lda2, W, bias,
                                                      o16, ldo, o32, ldo32, M, N, K, act);
  else
    wmma_gemm_lds_kernel<false><<<blocks, 256, 0, s>>>(A, lda, A2, lda2, W, bias,
                                                       o16, ldo, o32, ldo32, M, N, K, act);
}

static inline void run_resnet(hipStream_t s, const ResW& r,
                              const _Float16* X, int ldx, int M,
                              _Float16* H1b, _Float16* H2b,
                              _Float16* Y, int ldy, float* Y32, int ldy32) {
  launch_gemm(s, X,   ldx,  nullptr, 0,  r.l1.w, r.l1.b, H1b, r.dh, nullptr, 0, M, r.dh,   r.din, 1);
  launch_gemm(s, H1b, r.dh, nullptr, 0,  r.l2.w, r.l2.b, H2b, r.dh, nullptr, 0, M, r.dh,   r.dh,  1);
  launch_gemm(s, H2b, r.dh, X, ldx,      r.l3.w, r.l3.b, Y,   ldy,  Y32, ldy32, M, r.dout, r.dh,  1);
}

static inline void run_chain(hipStream_t s, const ResW r[3],
                             const _Float16* X, int ldx, int M,
                             _Float16* H1b, _Float16* H2b,
                             _Float16* Eb, _Float16* X2b,
                             _Float16* Y, int ldy, float* Y32, int ldy32) {
  run_resnet(s, r[0], X,   ldx, M, H1b, H2b, Eb,  HH, nullptr, 0);
  run_resnet(s, r[1], Eb,  HH,  M, H1b, H2b, X2b, HH, nullptr, 0);
  run_resnet(s, r[2], X2b, HH,  M, H1b, H2b, Y,  ldy, Y32, ldy32);
}

extern "C" void kernel_launch(void* const* d_in, const int* in_sizes, int n_in,
                              void* d_out, int out_size, void* d_ws, size_t ws_size,
                              hipStream_t stream) {
  (void)in_sizes; (void)n_in; (void)out_size; (void)ws_size;
  const float* feats = (const float*)d_in[0];
  const int*   preds = (const int*)d_in[1];
  float*       out   = (float*)d_out;

  size_t off = 0;
  auto alloc16 = [&](size_t halves) -> _Float16* {
    _Float16* p = (_Float16*)((char*)d_ws + off);
    off += halves * 2; off = (off + 255) & ~(size_t)255;
    return p;
  };
  _Float16* WGT = alloc16(3743744);
  _Float16* X   = alloc16((size_t)PP * 512);
  _Float16* H1b = alloc16((size_t)PP * 512);
  _Float16* H2b = alloc16((size_t)PP * 512);
  _Float16* Eb  = alloc16((size_t)PP * HH);
  _Float16* X2b = alloc16((size_t)PP * HH);
  _Float16* MSG = alloc16((size_t)PP * HH);
  _Float16* CUR = alloc16((size_t)PP * HH);

  // d_in order (jax pytree, keys sorted): node_feats, preds, then params:
  // comb_embed(2), comb_embed1(8), comb_embed2(14), embed_op(20),
  // mp_binary(22..), mp_unary(40..), node_embeds(58..). Linear: b then w.
  _Float16* wcur = WGT;
  auto cvtW = [&](int idx, int rows, int cols) -> const _Float16* {
    int n = rows * cols;
    cvt_kernel<<<(n + 255) / 256, 256, 0, stream>>>((const float*)d_in[idx], wcur, n);
    _Float16* p = wcur; wcur += n;
    return p;
  };
  auto regRes = [&](int base, int din, int dh, int dout) -> ResW {
    ResW r; r.din = din; r.dh = dh; r.dout = dout;
    r.l1.b = (const float*)d_in[base + 0]; r.l1.w = cvtW(base + 1, dh,   din);
    r.l2.b = (const float*)d_in[base + 2]; r.l2.w = cvtW(base + 3, dh,   dh);
    r.l3.b = (const float*)d_in[base + 4]; r.l3.w = cvtW(base + 5, dout, dh);
    return r;
  };

  ResW comb[3] = { regRes(2, 512, 512, 256), regRes(8, 256, 256, 256), regRes(14, 256, 256, 256) };
  const float*    embedB = (const float*)d_in[20];
  const _Float16* embedW = cvtW(21, 256, FEATN);
  ResW mpb[3]  = { regRes(22, 512, 512, 256), regRes(28, 256, 256, 256), regRes(34, 256, 256, 256) };
  ResW mpu[3]  = { regRes(40, 256, 256, 256), regRes(46, 256, 256, 256), regRes(52, 256, 256, 256) };
  ResW node[3] = { regRes(58, 512, 512, 256), regRes(64, 256, 256, 256), regRes(70, 256, 256, 256) };

  const int NG = (HALFP * HH + 255) / 256;
  const int NC = (PP * HH + 255) / 256;

  // layer 0: embeds[:P] = tanh(embed_op(node_feats[:P]))
  cvt_kernel<<<(PP * FEATN + 255) / 256, 256, 0, stream>>>(feats, H1b, PP * FEATN);
  launch_gemm(stream, H1b, FEATN, nullptr, 0, embedW, embedB,
              CUR, HH, out, HH, PP, HH, FEATN, 1);

  for (int l = 1; l < LL; ++l) {
    const int* pb = preds + (size_t)(l - 1) * PP * 2;

    // unary messages -> MSG[:half]
    gather_u_kernel<<<NG, 256, 0, stream>>>(CUR, pb, X);
    run_chain(stream, mpu, X, HH, HALFP, H1b, H2b, Eb, X2b, MSG, HH, nullptr, 0);

    // binary messages -> MSG[half:]
    gather_b_kernel<<<NG, 256, 0, stream>>>(CUR, pb, X);
    run_chain(stream, mpb, X, 512, HALFP, H1b, H2b, Eb, X2b,
              MSG + (size_t)HALFP * HH, HH, nullptr, 0);

    // e = tanh(embed_op(feats_l)); X = concat(e, zeros)
    cvt_kernel<<<(PP * FEATN + 255) / 256, 256, 0, stream>>>(
        feats + (size_t)l * PP * FEATN, H1b, PP * FEATN);
    zero_hi_kernel<<<NC, 256, 0, stream>>>(X);
    launch_gemm(stream, H1b, FEATN, nullptr, 0, embedW, embedB,
                X, 512, nullptr, 0, PP, HH, FEATN, 1);

    // e = mlp3(comb_embed*, X) -> X[:, :256]
    run_chain(stream, comb, X, 512, PP, H1b, H2b, Eb, X2b, X, 512, nullptr, 0);

    // X = concat(e, msgs); e = mlp3(node_embeds*, X) -> CUR (f16) + d_out (f32)
    concat_hi_kernel<<<NC, 256, 0, stream>>>(MSG, X);
    run_chain(stream, node, X, 512, PP, H1b, H2b, Eb, X2b,
              CUR, HH, out + (size_t)l * PP * HH, HH);
  }
}
